// EditModel_72301479461312
// MI455X (gfx1250) — matrix operand order repaired and checked by reading, not compiled, not run on hardware
//
#include <hip/hip_runtime.h>
#include <math.h>

// ---------------------------------------------------------------------------
// Problem constants (from reference): L=128, B=64, V=10000, E=512, H=512
// ---------------------------------------------------------------------------
#define LL 128
#define BB 64
#define VV 10000
#define EE 512
#define HH 512
#define G3 (3 * HH)       // 1536
#define LOUT (LL - 2)     // 126
#define MROWS (LL * BB)   // 8192
#define MDEC (LOUT * BB)  // 8064
#define KDEC (2 * HH)     // 1024

typedef __bf16 bf16_t;
typedef __attribute__((ext_vector_type(16))) __bf16 v16bf;
typedef __attribute__((ext_vector_type(8)))  float  v8f;
typedef int v4i __attribute__((vector_size(16)));   // matches builtin param type

#define AS1 __attribute__((address_space(1)))
#define AS3 __attribute__((address_space(3)))

// CDNA5 async global->LDS staging (ASYNCcnt-tracked), with sync fallback.
#if defined(__has_builtin)
#if __has_builtin(__builtin_amdgcn_global_load_async_to_lds_b128) && \
    __has_builtin(__builtin_amdgcn_s_wait_asynccnt)
#define USE_ASYNC_LDS 1
#endif
#endif
#ifndef USE_ASYNC_LDS
#define USE_ASYNC_LDS 0
#endif

__device__ __forceinline__ v8f wmma_bf16(v16bf a, v16bf b, v8f c) {
  // D = A(16x32) x B(32x16) + C, f32 accumulate
  return __builtin_amdgcn_wmma_f32_16x16x32_bf16(
      /*neg_a=*/false, a, /*neg_b=*/false, b,
      /*c_mod=*/(short)0, c, /*reuse_a=*/false, /*reuse_b=*/false);
}

// A fragment: 16x32 (MxK), row-major source with leading dim `ld` (elements).
// Lane layout (ISA 7.12.2, 16-bit A 16x32): lanes 0-15 -> M=lane, K chunks
// {0..7, 16..23}; lanes 16-31 -> M=lane-16, K chunks {8..15, 24..31}.
__device__ __forceinline__ v16bf frag_a(const bf16_t* p, int ld) {
  int lane = threadIdx.x & 31;
  const bf16_t* q = p + (size_t)(lane & 15) * ld + ((lane >> 4) << 3);
  v16bf f;
  reinterpret_cast<uint4*>(&f)[0] = *reinterpret_cast<const uint4*>(q);
  reinterpret_cast<uint4*>(&f)[1] = *reinterpret_cast<const uint4*>(q + 16);
  return f;
}

// B fragment: 32x16 (KxN). Source is row-major [N][K] (weight rows), leading
// dim `ld`. Lanes 0-15 -> N=lane, K {0..7, 16..23}; lanes 16-31 -> N=lane-16, K {8..15, 24..31}.
__device__ __forceinline__ v16bf frag_b(const bf16_t* p, int ld) {
  int lane = threadIdx.x & 31;
  const bf16_t* q = p + (size_t)(lane & 15) * ld + ((lane >> 4) << 3);
  v16bf f;
  reinterpret_cast<uint4*>(&f)[0] = *reinterpret_cast<const uint4*>(q);
  reinterpret_cast<uint4*>(&f)[1] = *reinterpret_cast<const uint4*>(q + 16);
  return f;
}

__device__ __forceinline__ float sigmoidf_(float x) {
  return 1.0f / (1.0f + __expf(-x));
}

// ---------------------------------------------------------------------------
// f32 -> bf16 conversion
// ---------------------------------------------------------------------------
__global__ __launch_bounds__(256) void cvt_f32_bf16_kernel(
    const float* __restrict__ src, bf16_t* __restrict__ dst, int n) {
  int i = blockIdx.x * 256 + threadIdx.x;
  if (i < n) dst[i] = (bf16_t)src[i];
}

// ---------------------------------------------------------------------------
// Embedding gather + cast to bf16. x_bf[(l*B+b)*E + e] = emb[seq[l*B+b]][e]
// ---------------------------------------------------------------------------
__global__ __launch_bounds__(256) void embed_kernel(
    const int* __restrict__ seq, const float* __restrict__ emb,
    bf16_t* __restrict__ xbf) {
  int gid = blockIdx.x * 256 + threadIdx.x;          // over MROWS * E/4
  int row = gid >> 7;                                // E/4 = 128 chunks/row
  int c4  = (gid & 127) << 2;
  if (row >= MROWS) return;
  int tok = seq[row];
  const float* s = emb + (size_t)tok * EE + c4;
  bf16_t* d = xbf + (size_t)row * EE + c4;
  float4 v = *reinterpret_cast<const float4*>(s);
  d[0] = (bf16_t)v.x; d[1] = (bf16_t)v.y; d[2] = (bf16_t)v.z; d[3] = (bf16_t)v.w;
}

// ---------------------------------------------------------------------------
// WMMA bf16 GEMM with fused bias, double-buffered LDS, async global->LDS:
//   C[M,N] (f32) = A[M,K] (bf16, row-major) * B[N,K]^T (bf16, row-major) + bias[N]
// Block tile 128x128, K-step 32. 8 waves: 4(M) x 2(N); wave tile 32x64
// (2x4 = 8 WMMA accumulators). Requires M % 128 == 0, K % 32 == 0. N guarded.
// ---------------------------------------------------------------------------
#define BM 128
#define BN 128
#define BK 32

__global__ __launch_bounds__(256) void wmma_gemm_bias_kernel(
    const bf16_t* __restrict__ A, const bf16_t* __restrict__ B,
    const float* __restrict__ bias, float* __restrict__ C,
    int M, int N, int K) {
  __shared__ __align__(16) bf16_t As[2][BM][BK];  // 2 x 8 KB
  __shared__ __align__(16) bf16_t Bs[2][BN][BK];  // 2 x 8 KB

  const int tid  = threadIdx.x;
  const int wave = tid >> 5;
  const int lane = tid & 31;
  const int m0 = blockIdx.y * BM;
  const int n0 = blockIdx.x * BN;
  const int wm = (wave >> 1) * 32;  // wave row offset: 0/32/64/96
  const int wn = (wave & 1) * 64;   // wave col offset: 0/64

  v8f acc[2][4] = {};

  // Stage one 128x32 A tile + 128x32 B tile into LDS buffer `buf`.
  // 512 16-byte chunks per tile; 256 threads x 2 chunks each.
  auto stage = [&](int buf, int k0) {
    #pragma unroll
    for (int i = 0; i < 2; ++i) {
      int c  = tid + i * 256;        // chunk id 0..511
      int r  = c >> 2;               // row 0..127
      int ks = (c & 3) << 3;         // 8-elem offset within 32-wide row
      const bf16_t* gsrc = A + (size_t)(m0 + r) * K + k0 + ks;
      bf16_t* ldst = &As[buf][r][ks];
#if USE_ASYNC_LDS
      __builtin_amdgcn_global_load_async_to_lds_b128(
          (AS1 v4i*)gsrc, (AS3 v4i*)ldst, 0, 0);
#else
      *reinterpret_cast<uint4*>(ldst) = *reinterpret_cast<const uint4*>(gsrc);
#endif
    }
    #pragma unroll
    for (int i = 0; i < 2; ++i) {
      int c  = tid + i * 256;
      int r  = c >> 2;
      int ks = (c & 3) << 3;
      int n  = n0 + r;
      bf16_t* ldst = &Bs[buf][r][ks];
      if (n < N) {
        const bf16_t* gsrc = B + (size_t)n * K + k0 + ks;
#if USE_ASYNC_LDS
        __builtin_amdgcn_global_load_async_to_lds_b128(
            (AS1 v4i*)gsrc, (AS3 v4i*)ldst, 0, 0);
#else
        *reinterpret_cast<uint4*>(ldst) = *reinterpret_cast<const uint4*>(gsrc);
#endif
      } else {
        uint4 z = {0u, 0u, 0u, 0u};
        *reinterpret_cast<uint4*>(ldst) = z;   // keep OOB rows finite
      }
    }
  };

  auto compute = [&](int buf) {
    v16bf af[2], bfr[4];
    #pragma unroll
    for (int mt = 0; mt < 2; ++mt) af[mt] = frag_a(&As[buf][wm + mt * 16][0], BK);
    #pragma unroll
    for (int nt = 0; nt < 4; ++nt) bfr[nt] = frag_b(&Bs[buf][wn + nt * 16][0], BK);
    #pragma unroll
    for (int mt = 0; mt < 2; ++mt)
      #pragma unroll
      for (int nt = 0; nt < 4; ++nt)
        acc[mt][nt] = wmma_bf16(af[mt], bfr[nt], acc[mt][nt]);
  };

  const int nk = K / BK;
  stage(0, 0);
  for (int kt = 0; kt < nk; ++kt) {
    const int p = kt & 1;
#if USE_ASYNC_LDS
    __builtin_amdgcn_s_wait_asynccnt(0);   // my async LDS writes done
#endif
    __syncthreads();                       // everyone's writes visible; prior
                                           // reads of buf[1-p] retired
    if (kt + 1 < nk) stage(1 - p, (kt + 1) * BK);
    compute(p);
  }

  // Store C (f32) + bias. D layout: element i -> row = 8*lane[4]+i, col = lane[3:0].
  #pragma unroll
  for (int mt = 0; mt < 2; ++mt) {
    #pragma unroll
    for (int nt = 0; nt < 4; ++nt) {
      int rbase = m0 + wm + mt * 16 + ((lane >> 4) << 3);
      int c     = n0 + wn + nt * 16 + (lane & 15);
      if (c < N) {
        float bv = bias[c];
        #pragma unroll
        for (int i = 0; i < 8; ++i)
          C[(size_t)(rbase + i) * N + c] = acc[mt][nt][i] + bv;
      }
    }
  }
}

// ---------------------------------------------------------------------------
// Zero-init h state (f32 + bf16) for both directions.
// ---------------------------------------------------------------------------
__global__ __launch_bounds__(256) void zero_h_kernel(
    float* hf, bf16_t* hfb, float* hb, bf16_t* hbb) {
  int i = blockIdx.x * 256 + threadIdx.x;
  if (i < BB * HH) {
    hf[i] = 0.0f; hfb[i] = (bf16_t)0.0f;
    hb[i] = 0.0f; hbb[i] = (bf16_t)0.0f;
  }
}

// ---------------------------------------------------------------------------
// One GRU timestep, both directions fused. Grid: 16 blocks (8 per direction),
// each block owns a 64-wide column slice j0..j0+63 of H and computes
// gh = h_prev @ W_hh^T for the 3 gate sections of that slice (WMMA), then
// the fused gate elementwise update. h is ping-ponged across launches.
// ---------------------------------------------------------------------------
__global__ __launch_bounds__(256) void gru_step_kernel(
    int t,
    const float* __restrict__ gx_f, const float* __restrict__ gx_b,
    const float* __restrict__ hf_in, const bf16_t* __restrict__ hf_in_bf,
    float* __restrict__ hf_out, bf16_t* __restrict__ hf_out_bf,
    const float* __restrict__ hb_in, const bf16_t* __restrict__ hb_in_bf,
    float* __restrict__ hb_out, bf16_t* __restrict__ hb_out_bf,
    const bf16_t* __restrict__ Whh_f_bf, const bf16_t* __restrict__ Whh_b_bf,
    const float* __restrict__ bhh_f, const float* __restrict__ bhh_b,
    float* __restrict__ out_f, float* __restrict__ out_b) {
  const int blocks_per_dir = HH / 64;  // 8
  const int dir = (blockIdx.x >= blocks_per_dir) ? 1 : 0;
  const int j0  = (blockIdx.x - dir * blocks_per_dir) * 64;

  const float*  gx      = dir ? gx_b : gx_f;
  const int     tstep   = dir ? (LL - 1 - t) : t;
  const float*  hin     = dir ? hb_in     : hf_in;
  const bf16_t* hin_bf  = dir ? hb_in_bf  : hf_in_bf;
  float*        hout    = dir ? hb_out    : hf_out;
  bf16_t*       hout_bf = dir ? hb_out_bf : hf_out_bf;
  const bf16_t* W       = dir ? Whh_b_bf  : Whh_f_bf;
  const float*  bhh     = dir ? bhh_b     : bhh_f;
  float*        outp    = dir ? out_b     : out_f;

  __shared__ float ghs[3][BB][64];  // 48 KB

  const int wave = threadIdx.x >> 5;
  const int lane = threadIdx.x & 31;

  // 48 (gate, m-tile, n-tile) 16x16 tiles, 8 waves -> 6 tiles each.
  for (int tt = wave; tt < 48; tt += 8) {
    int g   = tt >> 4;        // gate 0..2
    int sub = tt & 15;
    int mt  = sub >> 2;       // batch tile 0..3 (rows 16*mt)
    int nt  = sub & 3;        // col tile 0..3 (cols 16*nt within slice)
    v8f acc = {};
    const bf16_t* Arow = hin_bf + (size_t)(mt * 16) * HH;
    const bf16_t* Brow = W + (size_t)(g * HH + j0 + nt * 16) * HH;
    #pragma unroll 4
    for (int k0 = 0; k0 < HH; k0 += 32) {
      v16bf a = frag_a(Arow + k0, HH);
      v16bf b = frag_b(Brow + k0, HH);
      acc = wmma_bf16(a, b, acc);
    }
    int rbase = mt * 16 + ((lane >> 4) << 3);
    int c     = nt * 16 + (lane & 15);
    #pragma unroll
    for (int i = 0; i < 8; ++i)
      ghs[g][rbase + i][c] = acc[i];
  }
  __syncthreads();

  // Fused GRU gates for this column slice.
  for (int idx = threadIdx.x; idx < BB * 64; idx += 256) {
    int b  = idx >> 6;
    int jj = idx & 63;
    int j  = j0 + jj;
    const float* gxr = gx + ((size_t)tstep * BB + b) * G3;
    float ghr = ghs[0][b][jj] + bhh[j];
    float ghz = ghs[1][b][jj] + bhh[HH + j];
    float ghn = ghs[2][b][jj] + bhh[2 * HH + j];
    float r = sigmoidf_(gxr[j] + ghr);
    float z = sigmoidf_(gxr[HH + j] + ghz);
    float n = tanhf(gxr[2 * HH + j] + r * ghn);
    float hp = hin[b * HH + j];
    float hn = (1.0f - z) * n + z * hp;
    hout[b * HH + j]    = hn;
    hout_bf[b * HH + j] = (bf16_t)hn;
    outp[((size_t)tstep * BB + b) * HH + j] = hn;
  }
}

// ---------------------------------------------------------------------------
// Concat out_f[0:126] and out_b[2:128] -> bf16 [8064, 1024]
// ---------------------------------------------------------------------------
__global__ __launch_bounds__(256) void concat_bf16_kernel(
    const float* __restrict__ out_f, const float* __restrict__ out_b,
    bf16_t* __restrict__ cat) {
  size_t gid = (size_t)blockIdx.x * 256 + threadIdx.x;   // over MDEC * KDEC
  if (gid >= (size_t)MDEC * KDEC) return;
  int c   = (int)(gid & (KDEC - 1));
  int row = (int)(gid >> 10);           // KDEC = 1024
  int l = row / BB, b = row % BB;
  float v;
  if (c < HH) v = out_f[((size_t)l * BB + b) * HH + c];
  else        v = out_b[((size_t)(l + 2) * BB + b) * HH + (c - HH)];
  cat[gid] = (bf16_t)v;
}

// ---------------------------------------------------------------------------
// In-place row log-softmax over ncols (one block per row).
// ---------------------------------------------------------------------------
__global__ __launch_bounds__(256) void log_softmax_kernel(
    float* __restrict__ x, int ncols) {
  __shared__ float red[256];
  float* p = x + (size_t)blockIdx.x * ncols;

  float m = -INFINITY;
  for (int c = threadIdx.x; c < ncols; c += 256) m = fmaxf(m, p[c]);
  red[threadIdx.x] = m;
  __syncthreads();
  for (int s = 128; s > 0; s >>= 1) {
    if (threadIdx.x < s) red[threadIdx.x] = fmaxf(red[threadIdx.x], red[threadIdx.x + s]);
    __syncthreads();
  }
  m = red[0];
  __syncthreads();

  float sum = 0.0f;
  for (int c = threadIdx.x; c < ncols; c += 256) sum += __expf(p[c] - m);
  red[threadIdx.x] = sum;
  __syncthreads();
  for (int s = 128; s > 0; s >>= 1) {
    if (threadIdx.x < s) red[threadIdx.x] += red[threadIdx.x + s];
    __syncthreads();
  }
  float lse = m + __logf(red[0]);

  for (int c = threadIdx.x; c < ncols; c += 256) p[c] -= lse;
}

// ---------------------------------------------------------------------------
// Host launcher
// ---------------------------------------------------------------------------
extern "C" void kernel_launch(void* const* d_in, const int* in_sizes, int n_in,
                              void* d_out, int out_size, void* d_ws, size_t ws_size,
                              hipStream_t stream) {
  (void)in_sizes; (void)n_in; (void)out_size; (void)ws_size;

  const int*   seq    = (const int*)d_in[0];
  // d_in[1] = seq_length (all == L) -> pack/pad identity, unused
  const float* emb    = (const float*)d_in[2];
  const float* W_ih_f = (const float*)d_in[3];
  const float* W_hh_f = (const float*)d_in[4];
  const float* b_ih_f = (const float*)d_in[5];
  const float* b_hh_f = (const float*)d_in[6];
  const float* W_ih_b = (const float*)d_in[7];
  const float* W_hh_b = (const float*)d_in[8];
  const float* b_ih_b = (const float*)d_in[9];
  const float* b_hh_b = (const float*)d_in[10];
  const float* dec_W  = (const float*)d_in[11];
  const float* dec_b  = (const float*)d_in[12];
  float* out = (float*)d_out;   // [126, 64, 10000]

  // Workspace carve-out (bump allocator, 256B aligned)
  char* ws = (char*)d_ws;
  auto carve = [&](size_t bytes) -> char* {
    char* p = ws;
    ws += (bytes + 255) & ~((size_t)255);
    return p;
  };
  bf16_t* x_bf     = (bf16_t*)carve((size_t)MROWS * EE * sizeof(bf16_t));
  bf16_t* wih_f_bf = (bf16_t*)carve((size_t)G3 * EE * sizeof(bf16_t));
  bf16_t* wih_b_bf = (bf16_t*)carve((size_t)G3 * EE * sizeof(bf16_t));
  bf16_t* whh_f_bf = (bf16_t*)carve((size_t)G3 * HH * sizeof(bf16_t));
  bf16_t* whh_b_bf = (bf16_t*)carve((size_t)G3 * HH * sizeof(bf16_t));
  bf16_t* decw_bf  = (bf16_t*)carve((size_t)VV * KDEC * sizeof(bf16_t));
  float*  gx_f     = (float*)carve((size_t)MROWS * G3 * sizeof(float));
  float*  gx_b     = (float*)carve((size_t)MROWS * G3 * sizeof(float));
  float*  hbuf[4];
  bf16_t* hbuf_bf[4];
  for (int i = 0; i < 4; ++i) {
    hbuf[i]    = (float*)carve((size_t)BB * HH * sizeof(float));
    hbuf_bf[i] = (bf16_t*)carve((size_t)BB * HH * sizeof(bf16_t));
  }
  float*  out_f  = (float*)carve((size_t)LL * BB * HH * sizeof(float));
  float*  out_b  = (float*)carve((size_t)LL * BB * HH * sizeof(float));
  bf16_t* cat_bf = (bf16_t*)carve((size_t)MDEC * KDEC * sizeof(bf16_t));

  // 1) Weight conversions f32 -> bf16
  auto cvt = [&](const float* s, bf16_t* d, int n) {
    cvt_f32_bf16_kernel<<<(n + 255) / 256, 256, 0, stream>>>(s, d, n);
  };
  cvt(W_ih_f, wih_f_bf, G3 * EE);
  cvt(W_ih_b, wih_b_bf, G3 * EE);
  cvt(W_hh_f, whh_f_bf, G3 * HH);
  cvt(W_hh_b, whh_b_bf, G3 * HH);
  cvt(dec_W,  decw_bf,  VV * KDEC);

  // 2) Embedding gather -> bf16
  {
    int total = MROWS * (EE / 4);
    embed_kernel<<<(total + 255) / 256, 256, 0, stream>>>(seq, emb, x_bf);
  }

  // 3) gx = x @ W_ih^T + b_ih  (both directions)  [8192, 1536], K=512
  {
    dim3 grid(G3 / BN, MROWS / BM);
    wmma_gemm_bias_kernel<<<grid, 256, 0, stream>>>(x_bf, wih_f_bf, b_ih_f, gx_f,
                                                    MROWS, G3, EE);
    wmma_gemm_bias_kernel<<<grid, 256, 0, stream>>>(x_bf, wih_b_bf, b_ih_b, gx_b,
                                                    MROWS, G3, EE);
  }

  // 4) Recurrent scan, both directions fused, h ping-ponged across launches.
  zero_h_kernel<<<(BB * HH + 255) / 256, 256, 0, stream>>>(
      hbuf[0], hbuf_bf[0], hbuf[2], hbuf_bf[2]);
  for (int t = 0; t < LL; ++t) {
    int cur = t & 1;
    gru_step_kernel<<<16, 256, 0, stream>>>(
        t, gx_f, gx_b,
        hbuf[cur],     hbuf_bf[cur],     hbuf[1 - cur], hbuf_bf[1 - cur],
        hbuf[2 + cur], hbuf_bf[2 + cur], hbuf[3 - cur], hbuf_bf[3 - cur],
        whh_f_bf, whh_b_bf, b_hh_f, b_hh_b, out_f, out_b);
  }

  // 5) Concat fwd[0:126] / bwd[2:128] -> bf16 [8064, 1024]
  {
    size_t total = (size_t)MDEC * KDEC;
    concat_bf16_kernel<<<(int)((total + 255) / 256), 256, 0, stream>>>(out_f, out_b, cat_bf);
  }

  // 6) Decoder: logits = cat @ dec_W^T + dec_b -> d_out [8064, 10000]
  {
    dim3 grid((VV + BN - 1) / BN, MDEC / BM);
    wmma_gemm_bias_kernel<<<grid, 256, 0, stream>>>(cat_bf, decw_bf, dec_b, out,
                                                    MDEC, VV, KDEC);
  }

  // 7) In-place log-softmax over V per row
  log_softmax_kernel<<<MDEC, 256, 0, stream>>>(out, VV);
}
